// DNGPradianceField_66340064854265
// MI455X (gfx1250) — matrix operand order repaired and checked by reading, compiled, not run on hardware
//
#include <hip/hip_runtime.h>
#include <hip/hip_bf16.h>
#include <math.h>

typedef _Float16 half_t;
typedef __attribute__((ext_vector_type(16))) _Float16 v16h;
typedef __attribute__((ext_vector_type(8)))  _Float16 v8h;
typedef __attribute__((ext_vector_type(8)))  float    v8f;

#define TBL_MASK 0x7FFFFu            // T = 2^19
#define TBL      524288u
#define MOVING_STEP 0.000244140625f  // 1/4096
#define PI_F 3.14159265358979323846f

// Per-level grid scale/resolution (scale = 16*B^l - 1, B = 2^(8/15)); levels 0..4 are dense.
__constant__ float kScale[16] = {
  15.0f,          22.1563078f,   32.5134112f,   47.5029297f,   69.1967940f,
  100.5936680f,  146.0333894f,  211.7968367f,  306.9743776f,  444.7218839f,
  644.0795776f,  932.6038912f, 1350.1760900f, 1954.5160060f, 2829.1574140f, 4095.0f
};
__constant__ int kRes[16] = {16,24,34,49,71,102,148,213,308,446,646,934,1352,1956,2831,4096};

// ---------------- WMMA fragment helpers (CDNA5 wave32 layouts) ----------------
// A (16x32 f16): lane l holds row M=l&15; half e -> K = e + 8*(l>>4) + (e>=8 ? 8 : 0)
__device__ __forceinline__ v16h load_a_frag(const half_t* act, int stride, int row0,
                                            int kbase, int lane) {
  const int m  = row0 + (lane & 15);
  const int hi = (lane >> 4) & 1;
  const half_t* p = act + m * stride + kbase + hi * 8;
  v8h lo = *(const v8h*)(p);
  v8h hh = *(const v8h*)(p + 16);
  return __builtin_shufflevector(lo, hh, 0,1,2,3,4,5,6,7,8,9,10,11,12,13,14,15);
}
// B (32x16 f16): lane l holds col N=l&15; half e -> K = kbase + e + 8*(l>>4) + (e>=8 ? 8 : 0).
// Weights staged transposed in LDS as wT[N][Kpad] so this is two contiguous 8-half runs (p, p+16).
__device__ __forceinline__ v16h load_b_frag(const half_t* wT, int kpad, int ncol0,
                                            int kbase, int lane) {
  const int n  = ncol0 + (lane & 15);
  const int hi = (lane >> 4) & 1;
  const half_t* p = wT + n * kpad + kbase + hi * 8;
  v8h lo = *(const v8h*)(p);
  v8h hh = *(const v8h*)(p + 16);
  return __builtin_shufflevector(lo, hh, 0,1,2,3,4,5,6,7,8,9,10,11,12,13,14,15);
}
// C/D (16x16 f32): VGPR i, lane l -> M = row0 + i + 8*(l>>4), N = l&15
__device__ __forceinline__ void store_c_frag(half_t* act, int stride, int row0, int ncol0,
                                             v8f c, bool relu, int lane) {
  const int n  = ncol0 + (lane & 15);
  const int hi = (lane >> 4) & 1;
#pragma unroll
  for (int i = 0; i < 8; ++i) {
    float v = c[i];
    if (relu) v = fmaxf(v, 0.0f);
    act[(row0 + i + hi * 8) * stride + n] = (half_t)v;
  }
}

// One dense layer for a wave's 32 points, in place in `buf`.
// Input rows: stride KIN; output rows: stride NOUT. All A frags preloaded first.
template <int KIN, int NOUT, bool RELU>
__device__ __forceinline__ void layer(half_t* buf, const half_t* wT, int lane) {
  constexpr int KC = KIN / 32;
  constexpr int NT = NOUT / 16;
  v16h a[2][KC];
#pragma unroll
  for (int t = 0; t < 2; ++t)
#pragma unroll
    for (int kc = 0; kc < KC; ++kc)
      a[t][kc] = load_a_frag(buf, KIN, t * 16, kc * 32, lane);
  __builtin_amdgcn_wave_barrier();
#pragma unroll
  for (int t = 0; t < 2; ++t) {
#pragma unroll
    for (int nt = 0; nt < NT; ++nt) {
      v8f c;
#pragma unroll
      for (int i = 0; i < 8; ++i) c[i] = 0.0f;
#pragma unroll
      for (int kc = 0; kc < KC; ++kc) {
        v16h b = load_b_frag(wT, KIN, nt * 16, kc * 32, lane);
        c = __builtin_amdgcn_wmma_f32_16x16x32_f16(false, a[t][kc], false, b,
                                                   (short)0, c, false, false);
      }
      store_c_frag(buf, NOUT, t * 16, nt * 16, c, RELU, lane);
    }
  }
  __builtin_amdgcn_wave_barrier();
}

// Stage a global f32 weight matrix (K x N row-major) into LDS transposed f16 [N][Kpad].
__device__ __forceinline__ void stageT(const float* w, half_t* dst, int K, int N,
                                       int Kpad, int tid) {
  for (int i = tid; i < N * Kpad; i += 256) {
    int n = i / Kpad, k = i - n * Kpad;
    dst[i] = (k < K) ? (half_t)w[k * N + n] : (half_t)0.0f;
  }
}

__global__ __launch_bounds__(256)
void dngp_fused_kernel(const float* __restrict__ positions, const float* __restrict__ tvals,
                       const float* __restrict__ directions, const float* __restrict__ tables,
                       const float* __restrict__ w_wrap0, const float* __restrict__ w_wrap1,
                       const float* __restrict__ w_wrap2, const float* __restrict__ w_base0,
                       const float* __restrict__ w_base1, const float* __restrict__ w_head0,
                       const float* __restrict__ w_head1, const float* __restrict__ w_head2,
                       float* __restrict__ out, int N) {
  __shared__ __align__(16) half_t sW0[64 * 32];  // wrap0^T
  __shared__ __align__(16) half_t sW1[64 * 64];  // wrap1^T
  __shared__ __align__(16) half_t sB0[64 * 32];  // base0^T
  __shared__ __align__(16) half_t sB1[16 * 64];  // base1^T
  __shared__ __align__(16) half_t sH0[64 * 32];  // head0^T (K 31 -> padded 32)
  __shared__ __align__(16) half_t sH1[64 * 64];  // head1^T
  __shared__ float sW2[3 * 64];                  // wrap2^T (f32)
  __shared__ float sH2[3 * 64];                  // head2^T (f32)
  __shared__ __align__(16) half_t sBuf[8][32 * 64];

  const int tid = threadIdx.x;
  // ---- stage all weights once per block ----
  stageT(w_wrap0, sW0, 32, 64, 32, tid);
  stageT(w_wrap1, sW1, 64, 64, 64, tid);
  stageT(w_base0, sB0, 32, 64, 32, tid);
  stageT(w_base1, sB1, 64, 16, 64, tid);
  stageT(w_head0, sH0, 31, 64, 32, tid);
  stageT(w_head1, sH1, 64, 64, 64, tid);
  for (int i = tid; i < 3 * 64; i += 256) {
    int j = i >> 6, k = i & 63;
    sW2[i] = w_wrap2[k * 3 + j];
    sH2[i] = w_head2[k * 3 + j];
  }
  __syncthreads();

  const int lane = tid & 31;
  const int wave = tid >> 5;
  half_t* buf = sBuf[wave];
  const int pt  = blockIdx.x * 256 + wave * 32 + lane;
  const int ptc = pt < N ? pt : N - 1;  // clamp; keeps EXEC full for WMMA

  // ---- phase 1: frequency encoding of [x,y,z,t] (32 feats) ----
  float in4[4];
  in4[0] = positions[(size_t)ptc * 3 + 0];
  in4[1] = positions[(size_t)ptc * 3 + 1];
  in4[2] = positions[(size_t)ptc * 3 + 2];
  in4[3] = tvals[ptc];
  {
    half_t* row = buf + lane * 32;
#pragma unroll
    for (int d = 0; d < 4; ++d) {
#pragma unroll
      for (int j = 0; j < 4; ++j) {
        float ang = in4[d] * (PI_F * (float)(1 << j));
        row[d * 8 + j]     = (half_t)sinf(ang);
        row[d * 8 + 4 + j] = (half_t)cosf(ang);
      }
    }
  }
  __builtin_amdgcn_wave_barrier();

  // ---- phase 2: wrap MLP 32->64->64 (WMMA), then 64->3 scalar ----
  layer<32, 64, true>(buf, sW0, lane);
  layer<64, 64, true>(buf, sW1, lane);
  float o0 = 0.f, o1 = 0.f, o2 = 0.f;
  {
    const half_t* r = buf + lane * 64;
#pragma unroll
    for (int k = 0; k < 64; ++k) {
      float h = (float)r[k];
      o0 += h * sW2[k]; o1 += h * sW2[64 + k]; o2 += h * sW2[128 + k];
    }
  }
  float xn0 = (in4[0] + o0 * MOVING_STEP + 1.5f) * (1.0f / 3.0f);
  float xn1 = (in4[1] + o1 * MOVING_STEP + 1.5f) * (1.0f / 3.0f);
  float xn2 = (in4[2] + o2 * MOVING_STEP + 1.5f) * (1.0f / 3.0f);
  bool sel = (xn0 > 0.f) && (xn0 < 1.f) && (xn1 > 0.f) && (xn1 < 1.f) &&
             (xn2 > 0.f) && (xn2 < 1.f);
  __builtin_amdgcn_wave_barrier();

  // ---- phase 3: hash-grid encoding (16 levels x 2 feats); gathers hit L2 ----
  {
    half_t* row = buf + lane * 32;
    for (int l = 0; l < 16; ++l) {
      const float scale = kScale[l];
      const int   res   = kRes[l];
      const bool  dense = (l < 5);
      float px = xn0 * scale + 0.5f, py = xn1 * scale + 0.5f, pz = xn2 * scale + 0.5f;
      float fx = floorf(px), fy = floorf(py), fz = floorf(pz);
      float rx = px - fx, ry = py - fy, rz = pz - fz;
      int ix = (int)fx, iy = (int)fy, iz = (int)fz;
      const float* tab = tables + (size_t)l * TBL * 2u;
      float a0 = 0.f, a1 = 0.f;
#pragma unroll
      for (int c = 0; c < 8; ++c) {
        const int ci = (c >> 2) & 1, cj = (c >> 1) & 1, ck = c & 1;
        const int cx = ix + ci, cy = iy + cj, cz = iz + ck;
        unsigned idx;
        if (dense) {
          idx = (unsigned)(cx + res * (cy + res * cz));
        } else {
          idx = (((unsigned)cx) * 1u ^ ((unsigned)cy) * 2654435761u ^
                 ((unsigned)cz) * 805459861u) & TBL_MASK;
        }
        float w = (ci ? rx : 1.f - rx) * (cj ? ry : 1.f - ry) * (ck ? rz : 1.f - rz);
        float2 fv = *(const float2*)(tab + (size_t)idx * 2u);
        a0 += w * fv.x;
        a1 += w * fv.y;
      }
      row[2 * l]     = (half_t)a0;
      row[2 * l + 1] = (half_t)a1;
    }
  }
  __builtin_amdgcn_wave_barrier();

  // ---- phase 4: base MLP 32->64->16 (WMMA) ----
  layer<32, 64, true>(buf, sB0, lane);
  layer<64, 16, false>(buf, sB1, lane);
  float density, geo[15];
  {
    const half_t* r = buf + lane * 16;
    density = sel ? expf((float)r[0] - 1.0f) : 0.0f;
#pragma unroll
    for (int k = 0; k < 15; ++k) geo[k] = (float)r[1 + k];
  }
  __builtin_amdgcn_wave_barrier();

  // ---- phase 5: SH deg-4 encoding + concat with geo -> head input (32 feats) ----
  {
    float dx = directions[(size_t)ptc * 3 + 0];
    float dy = directions[(size_t)ptc * 3 + 1];
    float dz = directions[(size_t)ptc * 3 + 2];
    float inv = rsqrtf(dx * dx + dy * dy + dz * dz);
    float x = dx * inv, y = dy * inv, z = dz * inv;
    float xy = x * y, yz = y * z, xz = x * z;
    float x2 = x * x, y2 = y * y, z2 = z * z;
    float sh[16];
    sh[0]  = 0.28209479177387814f;
    sh[1]  = -0.48860251190291987f * y;
    sh[2]  =  0.48860251190291987f * z;
    sh[3]  = -0.48860251190291987f * x;
    sh[4]  =  1.0925484305920792f * xy;
    sh[5]  = -1.0925484305920792f * yz;
    sh[6]  =  0.94617469575756f * z2 - 0.31539156525252f;
    sh[7]  = -1.0925484305920792f * xz;
    sh[8]  =  0.5462742152960396f * (x2 - y2);
    sh[9]  =  0.5900435899266435f * y * (y2 - 3.0f * x2);
    sh[10] =  2.890611442640554f * xy * z;
    sh[11] =  0.4570457994644657f * y * (1.0f - 5.0f * z2);
    sh[12] =  0.3731763325901154f * z * (5.0f * z2 - 3.0f);
    sh[13] =  0.4570457994644657f * x * (1.0f - 5.0f * z2);
    sh[14] =  1.445305721320277f * z * (x2 - y2);
    sh[15] =  0.5900435899266435f * x * (3.0f * y2 - x2);
    half_t* row = buf + lane * 32;
#pragma unroll
    for (int i = 0; i < 16; ++i) row[i] = (half_t)sh[i];
#pragma unroll
    for (int k = 0; k < 15; ++k) row[16 + k] = (half_t)geo[k];
    row[31] = (half_t)0.0f;  // pad K=31 -> 32 (matches zero-padded sH0)
  }
  __builtin_amdgcn_wave_barrier();

  // ---- phase 6: head MLP 32->64->64 (WMMA), 64->3 scalar + sigmoid ----
  layer<32, 64, true>(buf, sH0, lane);
  layer<64, 64, true>(buf, sH1, lane);
  float s0 = 0.f, s1 = 0.f, s2 = 0.f;
  {
    const half_t* r = buf + lane * 64;
#pragma unroll
    for (int k = 0; k < 64; ++k) {
      float h = (float)r[k];
      s0 += h * sH2[k]; s1 += h * sH2[64 + k]; s2 += h * sH2[128 + k];
    }
  }
  if (pt < N) {
    out[(size_t)pt * 4 + 0] = 1.0f / (1.0f + expf(-s0));
    out[(size_t)pt * 4 + 1] = 1.0f / (1.0f + expf(-s1));
    out[(size_t)pt * 4 + 2] = 1.0f / (1.0f + expf(-s2));
    out[(size_t)pt * 4 + 3] = density;
  }
}

extern "C" void kernel_launch(void* const* d_in, const int* in_sizes, int n_in,
                              void* d_out, int out_size, void* d_ws, size_t ws_size,
                              hipStream_t stream) {
  const float* positions  = (const float*)d_in[0];
  const float* tvals      = (const float*)d_in[1];
  const float* directions = (const float*)d_in[2];
  const float* tables     = (const float*)d_in[3];
  const float* w_wrap0    = (const float*)d_in[4];
  const float* w_wrap1    = (const float*)d_in[5];
  const float* w_wrap2    = (const float*)d_in[6];
  const float* w_base0    = (const float*)d_in[7];
  const float* w_base1    = (const float*)d_in[8];
  const float* w_head0    = (const float*)d_in[9];
  const float* w_head1    = (const float*)d_in[10];
  const float* w_head2    = (const float*)d_in[11];
  float* out = (float*)d_out;
  const int N = in_sizes[1];            // t has N elements
  const int blocks = (N + 255) / 256;   // 256 threads = 8 waves, 32 pts/wave
  hipLaunchKernelGGL(dngp_fused_kernel, dim3(blocks), dim3(256), 0, stream,
                     positions, tvals, directions, tables,
                     w_wrap0, w_wrap1, w_wrap2, w_base0, w_base1,
                     w_head0, w_head1, w_head2, out, N);
}
